// Node_25847113188082
// MI455X (gfx1250) — matrix operand order repaired and checked by reading, not compiled, not run on hardware
//
#include <hip/hip_runtime.h>
#include <math.h>

typedef __attribute__((ext_vector_type(16))) _Float16 v16h;
typedef __attribute__((ext_vector_type(8)))  _Float16 v8h;
typedef __attribute__((ext_vector_type(8)))  float    v8f;

#define DEVI __device__ __forceinline__

// ---------------------------------------------------------------------------
// WMMA fragment loaders (CDNA5 ISA 7.12.2, 16-bit layouts, wave32)
// A (16x32, MxK): lane L holds row M=L%16; lanes 0-15 -> K {0..7} U {16..23},
//                 lanes 16-31 -> K {8..15} U {24..31}   (two 8-half chunks, +16 apart)
// B (32x16, KxN): lane L holds col N=L%16; lanes 0-15 -> K {0..7} U {16..23},
//                 lanes 16-31 -> K {8..15} U {24..31}   (same as A)
// C/D (16x16 f32): lane L holds col N=L%16; element r -> row (L<16 ? r : 8+r)
// ---------------------------------------------------------------------------
DEVI v16h ld_a16(const _Float16* p) {
  v8h lo = *(const v8h*)(p);
  v8h hi = *(const v8h*)(p + 16);
  return __builtin_shufflevector(lo, hi, 0,1,2,3,4,5,6,7,8,9,10,11,12,13,14,15);
}
DEVI v16h ld_b16(const _Float16* p) {
  v8h lo = *(const v8h*)(p);
  v8h hi = *(const v8h*)(p + 16);
  return __builtin_shufflevector(lo, hi, 0,1,2,3,4,5,6,7,8,9,10,11,12,13,14,15);
}
DEVI v8f wmma_f16(v16h a, v16h b, v8f c) {
  return __builtin_amdgcn_wmma_f32_16x16x32_f16(false, a, false, b, (short)0, c,
                                                false, false);
}
DEVI float gelu_f(float v) { return 0.5f * v * (1.0f + erff(v * 0.70710678118654752f)); }

// ---------------------------------------------------------------------------
// Generic NT GEMM: out[M,N] = epi(A[M,K] @ W[N,K]^T + bias)
// one wave per block; wave tile = (MT*16) x (NT*16) = 64 x 64
// per K-step: 16 WMMAs vs 16 b128 loads (1.0 loads/WMMA)
// EPI: 0=bias->f32, 1=gelu->f32, 2=gelu->f16, 3=relu->f16,
//      4=sigmoid(.)*aux->f32, 5=bias+aux->f32
// ---------------------------------------------------------------------------
template<int MT, int NT, int EPI>
__global__ __launch_bounds__(32)
void gemm_nt(const _Float16* __restrict__ A, const _Float16* __restrict__ W,
             const float* __restrict__ bias, const float* __restrict__ aux,
             float* __restrict__ outF, _Float16* __restrict__ outH,
             int M, int N, int K)
{
  const int lane  = threadIdx.x & 31;
  const int row   = lane & 15;
  const int kofsA = (lane >> 4) << 3;   // 0 or 8
  const int kofsB = (lane >> 4) << 3;   // 0 or 8
  const int m0 = blockIdx.y * (MT * 16);
  const int n0 = blockIdx.x * (NT * 16);

  v8f acc[MT][NT] = {};
  const _Float16* Ap = A + (size_t)(m0 + row) * K + kofsA;
  const _Float16* Wp = W + (size_t)(n0 + row) * K + kofsB;

  for (int k = 0; k < K; k += 32) {
    v16h af[MT], bf[NT];
#pragma unroll
    for (int i = 0; i < MT; i++) af[i] = ld_a16(Ap + (size_t)i * 16 * K + k);
#pragma unroll
    for (int j = 0; j < NT; j++) bf[j] = ld_b16(Wp + (size_t)j * 16 * K + k);
#pragma unroll
    for (int i = 0; i < MT; i++)
#pragma unroll
      for (int j = 0; j < NT; j++)
        acc[i][j] = wmma_f16(af[i], bf[j], acc[i][j]);
  }

  const int col = lane & 15;
  const int rb  = (lane >> 4) << 3;
#pragma unroll
  for (int i = 0; i < MT; i++)
#pragma unroll
    for (int j = 0; j < NT; j++)
#pragma unroll
      for (int r = 0; r < 8; r++) {
        int m = m0 + i * 16 + rb + r;
        int n = n0 + j * 16 + col;
        size_t o = (size_t)m * N + n;
        float v = acc[i][j][r] + bias[n];
        if      (EPI == 0) outF[o] = v;
        else if (EPI == 1) outF[o] = gelu_f(v);
        else if (EPI == 2) outH[o] = (_Float16)gelu_f(v);
        else if (EPI == 3) outH[o] = (_Float16)fmaxf(v, 0.0f);
        else if (EPI == 4) { float s = 1.0f / (1.0f + expf(-v)); outF[o] = s * aux[o]; }
        else               outF[o] = v + aux[o];
      }
}

// ---------------------------------------------------------------------------
// Pointwise kernels
// ---------------------------------------------------------------------------
__global__ void prep_x_kernel(const float* __restrict__ x_list,
                              float* __restrict__ x, _Float16* __restrict__ xh,
                              size_t n) {
  size_t i = (size_t)blockIdx.x * blockDim.x + threadIdx.x;
  if (i < n) { float v = x_list[i * 2 + 1]; x[i] = v; xh[i] = (_Float16)v; }
}

__global__ void f32_to_f16_kernel(const float* __restrict__ src,
                                  _Float16* __restrict__ dst, size_t n) {
  size_t i = (size_t)blockIdx.x * blockDim.x + threadIdx.x;
  if (i < n) dst[i] = (_Float16)src[i];
}

__global__ void zero_f32_kernel(float* __restrict__ p, int n) {
  int i = blockIdx.x * blockDim.x + threadIdx.x;
  if (i < n) p[i] = 0.0f;
}

// weighted mix: xc = [x0, x1, op0, op1, x, op3]; w = softmax(w_/0.01)
__global__ void mix_kernel(const float* __restrict__ x_list, const float* __restrict__ w_,
                           const float* __restrict__ op0, const float* __restrict__ op1,
                           const float* __restrict__ x,   const float* __restrict__ op3,
                           float* __restrict__ xmix, _Float16* __restrict__ xmixh,
                           size_t n) {
  size_t i = (size_t)blockIdx.x * blockDim.x + threadIdx.x;
  if (i >= n) return;
  float wl[6]; float mx = -1e30f;
  for (int j = 0; j < 6; j++) { wl[j] = w_[j] * 100.0f; mx = fmaxf(mx, wl[j]); }
  float s = 0.0f;
  for (int j = 0; j < 6; j++) { wl[j] = expf(wl[j] - mx); s += wl[j]; }
  float inv = 1.0f / s;
  float v = (wl[0] * x_list[i * 2] + wl[1] * x_list[i * 2 + 1] +
             wl[2] * op0[i] + wl[3] * op1[i] + wl[4] * x[i] + wl[5] * op3[i]) * inv;
  xmix[i] = v; xmixh[i] = (_Float16)v;
}

// per-(token, head) LayerNorm on q,k (d=64) + f16 packing; v transposed to [d, n]
__global__ void ln_qkv_kernel(const float* __restrict__ qkv,
                              const float* __restrict__ nq_g, const float* __restrict__ nq_b,
                              const float* __restrict__ nk_g, const float* __restrict__ nk_b,
                              _Float16* __restrict__ q_h, _Float16* __restrict__ k_h,
                              _Float16* __restrict__ vt_h) {
  int idx = blockIdx.x * blockDim.x + threadIdx.x;   // 4096*16
  int tok = idx >> 4;
  int h   = idx & 15;
  int b   = tok >> 10;
  int n   = tok & 1023;
  int bh  = b * 16 + h;
  const float* qp = qkv + (size_t)tok * 3072 + h * 64;
  const float* kp = qp + 1024;
  const float* vp = qp + 2048;
  size_t qb = ((size_t)bh * 1024 + n) * 64;

  float s = 0.0f, ss = 0.0f;
  for (int d = 0; d < 64; d++) { float v = qp[d]; s += v; ss += v * v; }
  float mean = s * 0.015625f;
  float rstd = rsqrtf(ss * 0.015625f - mean * mean + 1e-5f);
  for (int d = 0; d < 64; d++)
    q_h[qb + d] = (_Float16)(((qp[d] - mean) * rstd) * nq_g[d] + nq_b[d]);

  s = 0.0f; ss = 0.0f;
  for (int d = 0; d < 64; d++) { float v = kp[d]; s += v; ss += v * v; }
  mean = s * 0.015625f;
  rstd = rsqrtf(ss * 0.015625f - mean * mean + 1e-5f);
  for (int d = 0; d < 64; d++)
    k_h[qb + d] = (_Float16)(((kp[d] - mean) * rstd) * nk_g[d] + nk_b[d]);

  size_t vb = (size_t)bh * 64 * 1024 + n;
  for (int d = 0; d < 64; d++) vt_h[vb + (size_t)d * 1024] = (_Float16)vp[d];
}

// ---------------------------------------------------------------------------
// Attention: one wave per (b,h, 16-query block). 32KB LDS score row-block.
//   A: S = (Q Kt)/8 via WMMA -> f16 LDS      B: row softmax (+attn_map row 0)
//   C: O = P V via WMMA (A-fragments from LDS)
// ---------------------------------------------------------------------------
__global__ __launch_bounds__(32)
void attn_kernel(const _Float16* __restrict__ q_h, const _Float16* __restrict__ k_h,
                 const _Float16* __restrict__ vt_h, _Float16* __restrict__ attn_h,
                 float* __restrict__ attn_map) {
  __shared__ _Float16 P[16 * 1024];
  const int lane  = threadIdx.x & 31;
  const int row   = lane & 15;
  const int kofsA = (lane >> 4) << 3;
  const int kofsB = (lane >> 4) << 3;
  const int bh = blockIdx.y;               // 0..63
  const int b  = bh >> 4, h = bh & 15;
  const int q0 = blockIdx.x * 16;          // 0..1008

  const _Float16* Q  = q_h  + (size_t)bh * 1024 * 64;
  const _Float16* Km = k_h  + (size_t)bh * 1024 * 64;
  const _Float16* Vt = vt_h + (size_t)bh * 64 * 1024;

  // Q fragments (16x64 -> two K-steps), kept in registers
  v16h aq0 = ld_a16(Q + (size_t)(q0 + row) * 64 +  0 + kofsA);
  v16h aq1 = ld_a16(Q + (size_t)(q0 + row) * 64 + 32 + kofsA);

  // phase A: scores -> LDS (f16, pre-scaled by 1/sqrt(64))
  const int col = lane & 15;
  const int rb  = (lane >> 4) << 3;
  for (int nb = 0; nb < 1024; nb += 16) {
    v16h bk0 = ld_b16(Km + (size_t)(nb + row) * 64 +  0 + kofsB);
    v16h bk1 = ld_b16(Km + (size_t)(nb + row) * 64 + 32 + kofsB);
    v8f s = {};
    s = wmma_f16(aq0, bk0, s);
    s = wmma_f16(aq1, bk1, s);
#pragma unroll
    for (int r = 0; r < 8; r++)
      P[(rb + r) * 1024 + nb + col] = (_Float16)(s[r] * 0.125f);
  }
  __syncthreads();

  // phase B: softmax per row; lane pair (L, L+16) splits the 1024 columns;
  // vectorized v8h (b128) LDS traffic
  {
    int r = lane & 15;
    int half = lane >> 4;
    _Float16* Prow = P + r * 1024 + half * 512;
    float mx = -1e30f;
    for (int i = 0; i < 64; i++) {
      v8h c = *(const v8h*)(Prow + i * 8);
#pragma unroll
      for (int j = 0; j < 8; j++) mx = fmaxf(mx, (float)c[j]);
    }
    mx = fmaxf(mx, __shfl_xor(mx, 16));
    float sum = 0.0f;
    for (int i = 0; i < 64; i++) {
      v8h c = *(const v8h*)(Prow + i * 8);
#pragma unroll
      for (int j = 0; j < 8; j++) sum += expf((float)c[j] - mx);
    }
    sum += __shfl_xor(sum, 16);
    float inv = 1.0f / sum;
    bool do_map = (blockIdx.x == 0) && (r == 0);
    for (int i = 0; i < 64; i++) {
      v8h c = *(const v8h*)(Prow + i * 8);
      v8h pv;
#pragma unroll
      for (int j = 0; j < 8; j++) {
        float p = expf((float)c[j] - mx) * inv;
        pv[j] = (_Float16)p;
        if (do_map) {
          int m = half * 512 + i * 8 + j;
          if (m >= 1) atomicAdd(&attn_map[b * 1023 + (m - 1)], p * 0.0625f);
        }
      }
      *(v8h*)(Prow + i * 8) = pv;
    }
  }
  __syncthreads();

  // phase C: O[16x64] = P[16x1024] @ V[1024x64]  (B = Vt rows)
  v8f o[4] = {};
  for (int k = 0; k < 1024; k += 32) {
    v16h ap = ld_a16(&P[row * 1024 + k + kofsA]);
#pragma unroll
    for (int j = 0; j < 4; j++) {
      v16h bv = ld_b16(Vt + (size_t)(j * 16 + row) * 1024 + k + kofsB);
      o[j] = wmma_f16(ap, bv, o[j]);
    }
  }
#pragma unroll
  for (int j = 0; j < 4; j++)
#pragma unroll
    for (int r = 0; r < 8; r++) {
      int m = q0 + rb + r;
      int n = h * 64 + j * 16 + col;
      attn_h[((size_t)(b * 1024 + m)) * 1024 + n] = (_Float16)o[j][r];
    }
}

// ---------------------------------------------------------------------------
// Host orchestration
// ---------------------------------------------------------------------------
extern "C" void kernel_launch(void* const* d_in, const int* in_sizes, int n_in,
                              void* d_out, int out_size, void* d_ws, size_t ws_size,
                              hipStream_t stream) {
  const float* x_list = (const float*)d_in[0];
  const float* w_     = (const float*)d_in[1];
  const float* qkv_w  = (const float*)d_in[2];
  const float* qkv_b  = (const float*)d_in[3];
  const float* proj_w = (const float*)d_in[4];
  const float* proj_b = (const float*)d_in[5];
  const float* nq_g   = (const float*)d_in[6];
  const float* nq_b   = (const float*)d_in[7];
  const float* nk_g   = (const float*)d_in[8];
  const float* nk_b   = (const float*)d_in[9];
  const float* conv_w = (const float*)d_in[10];
  const float* conv_b = (const float*)d_in[11];
  const float* ca1_w  = (const float*)d_in[12];
  const float* ca1_b  = (const float*)d_in[13];
  const float* ca2_w  = (const float*)d_in[14];
  const float* ca2_b  = (const float*)d_in[15];
  const float* mlp1_w = (const float*)d_in[16];
  const float* mlp1_b = (const float*)d_in[17];
  const float* mlp2_w = (const float*)d_in[18];
  const float* mlp2_b = (const float*)d_in[19];

  float* outF     = (float*)d_out;                       // [4,1024,1024]
  float* attn_map = outF + (size_t)4 * 1024 * 1024;      // [4,1023]

  const size_t TOK = 4096, C = 1024;
  char* ws = (char*)d_ws; size_t off = 0;
  auto alloc = [&](size_t bytes) -> char* {
    char* p = ws + off; off += (bytes + 255) & ~(size_t)255; return p;
  };

  float*     xf    = (float*)    alloc(TOK * C * 4);
  _Float16*  xh    = (_Float16*) alloc(TOK * C * 2);
  _Float16*  Hh    = (_Float16*) alloc(TOK * 64 * 2);
  float*     xmix  = (float*)    alloc(TOK * C * 4);
  _Float16*  xmixh = (_Float16*) alloc(TOK * C * 2);
  // union1: mlp hidden (f16, 32MB) then qkv activations (f32, 48MB)
  char* u1 = alloc((size_t)TOK * 3072 * 4);
  _Float16* Mh   = (_Float16*)u1;
  float*    qkvf = (float*)u1;
  // union2: op0/op1/op3 (f32, 48MB) then q/k/vt/attn f16 (32MB)
  char* u2 = alloc((size_t)3 * TOK * C * 4);
  float* op0 = (float*)u2;
  float* op1 = op0 + TOK * C;
  float* op3 = op1 + TOK * C;
  _Float16* qh  = (_Float16*)u2;
  _Float16* kh  = qh  + TOK * C;
  _Float16* vth = kh  + TOK * C;
  _Float16* ah  = vth + TOK * C;
  // f16 weight copies
  _Float16* whc  = (_Float16*)alloc((size_t)C * C * 2);
  _Float16* whc1 = (_Float16*)alloc((size_t)64 * C * 2);
  _Float16* whc2 = (_Float16*)alloc((size_t)C * 64 * 2);
  _Float16* whm1 = (_Float16*)alloc((size_t)4 * C * C * 2);
  _Float16* whm2 = (_Float16*)alloc((size_t)C * 4 * C * 2);
  _Float16* whq  = (_Float16*)alloc((size_t)3 * C * C * 2);
  _Float16* whp  = (_Float16*)alloc((size_t)C * C * 2);

  auto cvt = [&](const float* s, _Float16* d, size_t n) {
    f32_to_f16_kernel<<<dim3((unsigned)((n + 255) / 256)), 256, 0, stream>>>(s, d, n);
  };

  prep_x_kernel<<<dim3((unsigned)(TOK * C / 256)), 256, 0, stream>>>(x_list, xf, xh, TOK * C);
  cvt(conv_w, whc,  (size_t)C * C);
  cvt(ca1_w,  whc1, (size_t)64 * C);
  cvt(ca2_w,  whc2, (size_t)C * 64);
  cvt(mlp1_w, whm1, (size_t)4 * C * C);
  cvt(mlp2_w, whm2, (size_t)C * 4 * C);
  cvt(qkv_w,  whq,  (size_t)3 * C * C);
  cvt(proj_w, whp,  (size_t)C * C);

  // op0 = gelu(x @ conv_w^T + b)
  gemm_nt<4, 4, 1><<<dim3(16, 64), 32, 0, stream>>>(xh, whc, conv_b, nullptr, op0, nullptr, 4096, 1024, 1024);
  // H = relu(x @ ca1_w^T + b)  [4096 x 64] f16
  gemm_nt<4, 4, 3><<<dim3(1, 64), 32, 0, stream>>>(xh, whc1, ca1_b, nullptr, nullptr, Hh, 4096, 64, 1024);
  // op1 = sigmoid(H @ ca2_w^T + b) * x
  gemm_nt<4, 4, 4><<<dim3(16, 64), 32, 0, stream>>>(Hh, whc2, ca2_b, xf, op1, nullptr, 4096, 1024, 64);
  // M = gelu(x @ mlp1_w^T + b)  [4096 x 4096] f16
  gemm_nt<4, 4, 2><<<dim3(64, 64), 32, 0, stream>>>(xh, whm1, mlp1_b, nullptr, nullptr, Mh, 4096, 4096, 1024);
  // op3 = M @ mlp2_w^T + b
  gemm_nt<4, 4, 0><<<dim3(16, 64), 32, 0, stream>>>(Mh, whm2, mlp2_b, nullptr, op3, nullptr, 4096, 1024, 4096);
  // softmax-weighted mix
  mix_kernel<<<dim3((unsigned)(TOK * C / 256)), 256, 0, stream>>>(x_list, w_, op0, op1, xf, op3, xmix, xmixh, TOK * C);
  // attn_map must start at zero (atomic accumulation)
  zero_f32_kernel<<<dim3(16), 256, 0, stream>>>(attn_map, 4 * 1023);
  // qkv = xmix @ qkv_w^T + b   [4096 x 3072]  (overwrites Mh region - dead)
  gemm_nt<4, 4, 0><<<dim3(48, 64), 32, 0, stream>>>(xmixh, whq, qkv_b, nullptr, qkvf, nullptr, 4096, 3072, 1024);
  // per-head LN + f16 pack + V transpose (overwrites op0/op1/op3 - dead)
  ln_qkv_kernel<<<dim3(256), 256, 0, stream>>>(qkvf, nq_g, nq_b, nk_g, nk_b, qh, kh, vth);
  // attention
  attn_kernel<<<dim3(64, 64), 32, 0, stream>>>(qh, kh, vth, ah, attn_map);
  // out = attn @ proj_w^T + b + xmix
  gemm_nt<4, 4, 5><<<dim3(16, 64), 32, 0, stream>>>(ah, whp, proj_b, xmix, outF, nullptr, 4096, 1024, 1024);
}
